// GCLSTMModel_28630251995698
// MI455X (gfx1250) — compile-verified, matrix operand order checked by reading, not run on hardware
//
#include <hip/hip_runtime.h>
#include <hip/hip_bf16.h>

typedef __attribute__((ext_vector_type(16))) __bf16        v16bf;
typedef __attribute__((ext_vector_type(8)))  __bf16        v8bf;
typedef __attribute__((ext_vector_type(8)))  float         v8f;
typedef __attribute__((ext_vector_type(2)))  float         v2f;
typedef __attribute__((ext_vector_type(4)))  unsigned int  uint4v;
typedef __attribute__((ext_vector_type(8)))  int           int8v;
typedef __attribute__((ext_vector_type(4)))  int           int4v;

#define N_ST   512
#define TSTEPS 168
#define FDIM   8
#define HDIM   64

// Load a 16-lane-striped WMMA operand chunk (A layout, 16-bit, K=32 step):
// lane L: row = L&15 (caller adds row), cols kbase..kbase+7 and kbase+16..kbase+23
// where kbase = (L>>4)*8. base0 must already point at row*stride + k0 + (L>>4)*8.
__device__ __forceinline__ v16bf load_tile16(const __bf16* base0) {
    v8bf lo = *(const v8bf*)(base0);
    v8bf hi = *(const v8bf*)(base0 + 16);
    return __builtin_shufflevector(lo, hi, 0,1,2,3,4,5,6,7,8,9,10,11,12,13,14,15);
}

__device__ __forceinline__ float sigm(float x) {
    return 1.0f / (1.0f + __expf(-x));
}
__device__ __forceinline__ float fast_tanh(float x) {
    const float e = __expf(2.0f * x);
    return 1.0f - 2.0f / (e + 1.0f);     // saturates correctly at +/-1
}

__global__ __launch_bounds__(256, 1)
void gclstm_kernel(const float*  __restrict__ seq,   // (32, 512, 168, 8) f32
                   const __bf16* __restrict__ adjb,  // (512, 512) bf16 row-major
                   const float*  __restrict__ Wg,    // (72, 64)
                   const float*  __restrict__ bg,    // (64)
                   const float*  __restrict__ Wl,    // (64, 256)
                   const float*  __restrict__ bl,    // (256)
                   const float*  __restrict__ Wr1,   // (64, 32)
                   const float*  __restrict__ br1,   // (32)
                   const float*  __restrict__ Wr2,   // (32, 1)
                   const float*  __restrict__ br2,   // (1)
                   float*        __restrict__ out)   // (32)
{
    // ---------------- LDS (~315 KB) ----------------
    __shared__ __bf16 h_s[N_ST * HDIM];          // 64 KB  hidden state (bf16, WMMA A)
    __shared__ float  c_s[N_ST * HDIM];          // 128 KB cell state (f32)
    __shared__ float  xagg_s[N_ST * FDIM];       // 16 KB  x_t; aliased as per-wave agg scratch
    __shared__ __bf16 msgT_s[HDIM * N_ST];       // 64 KB  msg transposed [hdim][station]
    __shared__ __bf16 WghT_s[HDIM * HDIM];       // 8 KB   Wg(h-part) transposed [out][in]
    __shared__ float  WgxT_s[HDIM * FDIM];       // 2 KB   Wg(x-part) transposed [out][in]
    __shared__ __bf16 WlT_s[4 * HDIM * HDIM];    // 32 KB  Wl transposed [out 256][in 64]
    __shared__ float  bg_s[HDIM];
    __shared__ float  bl_s[4 * HDIM];

    const int b     = blockIdx.x;
    const int tid   = threadIdx.x;
    const int lane  = tid & 31;
    const int wave  = tid >> 5;
    const int lrow  = lane & 15;   // tile row (A) or tile col (B/C)
    const int lhalf = lane >> 4;   // 0/1 half selector
    const int m0w   = wave * 64;   // this wave owns stations [m0w, m0w+64)

    // ---------------- one-time staging ----------------
    for (int i = tid; i < HDIM * HDIM; i += 256) {           // h-part of Wg, transposed
        int o = i / HDIM, k = i % HDIM;
        WghT_s[o * HDIM + k] = (__bf16)Wg[(FDIM + k) * HDIM + o];
    }
    for (int i = tid; i < HDIM * FDIM; i += 256) {           // x-part of Wg, transposed
        int o = i / FDIM, k = i % FDIM;
        WgxT_s[i] = Wg[k * HDIM + o];
    }
    for (int i = tid; i < 4 * HDIM * HDIM; i += 256) {       // Wl transposed
        int o = i / HDIM, k = i % HDIM;
        WlT_s[o * HDIM + k] = (__bf16)Wl[k * (4 * HDIM) + o];
    }
    for (int i = tid; i < HDIM; i += 256)     bg_s[i] = bg[i];
    for (int i = tid; i < 4 * HDIM; i += 256) bl_s[i] = bl[i];
    for (int i = tid; i < N_ST * HDIM; i += 256) {           // h0 = c0 = 0
        h_s[i] = (__bf16)0.0f;
        c_s[i] = 0.0f;
    }
    __syncthreads();

    const float* seq_b = seq + (size_t)b * N_ST * TSTEPS * FDIM;
    __bf16* aggscr = (__bf16*)xagg_s + wave * (16 * HDIM);   // 2 KB/wave, == wave's x region

    for (int t = 0; t < TSTEPS; ++t) {
        // ---- stage x_t via Tensor Data Mover: 64x8 f32 tile -> this wave's LDS region ----
        {
            const float*   gsrc     = seq_b + (size_t)m0w * TSTEPS * FDIM + (size_t)t * FDIM;
            const unsigned long long ga = (unsigned long long)(uintptr_t)gsrc;
            const unsigned int lds_base = (unsigned int)(uintptr_t)(xagg_s + m0w * FDIM);

            uint4v g0;
            g0[0] = 1u;                                       // count=1 (valid descriptor)
            g0[1] = lds_base;                                 // lds_addr (bytes)
            g0[2] = (unsigned int)ga;                         // global_addr[31:0]
            g0[3] = (unsigned int)(ga >> 32) | (2u << 30);    // global_addr[56:32] | type=2

            int8v g1;
            g1[0] = 0x00020000;                               // data_size=4B, mask=0
            g1[1] = (int)(8u  << 16);                         // tensor_dim0 = 8
            g1[2] = (int)(64u << 16);                         // tensor_dim0_hi=0 | tensor_dim1 = 64
            g1[3] = (int)(8u  << 16);                         // tensor_dim1_hi=0 | tile_dim0 = 8
            g1[4] = 64;                                       // tile_dim1 = 64, tile_dim2 = 0
            g1[5] = TSTEPS * FDIM;                            // tensor_dim0_stride = 1344
            g1[6] = 0;
            g1[7] = 0;

            int4v gz = {0, 0, 0, 0};
#if __clang_major__ >= 23
            int8v gz8 = {0, 0, 0, 0, 0, 0, 0, 0};
            __builtin_amdgcn_tensor_load_to_lds(g0, g1, gz, gz, gz8, 0);
#else
            __builtin_amdgcn_tensor_load_to_lds(g0, g1, gz, gz, 0);
#endif
        }

        // ================= GEMM1: msg = relu(h@Wgh + x@Wgx + bg) =================
        for (int mt = 0; mt < 4; ++mt) {
            const int m0 = m0w + mt * 16;
            v8f acc[4];
            #pragma unroll
            for (int nt = 0; nt < 4; ++nt) {
                const float bv = bg_s[nt * 16 + lrow];
                #pragma unroll
                for (int r = 0; r < 8; ++r) acc[nt][r] = bv;
            }
            // h @ Wgh : K=64, two bf16 WMMA k-steps
            #pragma unroll
            for (int ks = 0; ks < 2; ++ks) {
                v16bf A = load_tile16(h_s + (m0 + lrow) * HDIM + ks * 32 + lhalf * 8);
                #pragma unroll
                for (int nt = 0; nt < 4; ++nt) {
                    v16bf B = load_tile16(WghT_s + (nt * 16 + lrow) * HDIM + ks * 32 + lhalf * 8);
                    acc[nt] = __builtin_amdgcn_wmma_f32_16x16x32_bf16(
                        false, A, false, B, (short)0, acc[nt], false, false);
                }
            }
            // x @ Wgx : K=8 via two f32 WMMA k-steps (16x16x4), exact f32 path
            __builtin_amdgcn_s_wait_tensorcnt((short)0);      // TDM tile landed
            #pragma unroll
            for (int ks = 0; ks < 2; ++ks) {
                v2f A2 = *(const v2f*)(xagg_s + (m0 + lrow) * FDIM + ks * 4 + lhalf * 2);
                #pragma unroll
                for (int nt = 0; nt < 4; ++nt) {
                    v2f B2 = *(const v2f*)(WgxT_s + (nt * 16 + lrow) * FDIM + ks * 4 + lhalf * 2);
                    acc[nt] = __builtin_amdgcn_wmma_f32_16x16x4_f32(
                        false, A2, false, B2, (short)0, acc[nt], false, false);
                }
            }
            // relu + store transposed into msgT
            #pragma unroll
            for (int nt = 0; nt < 4; ++nt) {
                const int col = nt * 16 + lrow;
                #pragma unroll
                for (int r = 0; r < 8; ++r) {
                    float v = acc[nt][r];
                    v = v > 0.0f ? v : 0.0f;
                    msgT_s[col * N_ST + (m0 + r + lhalf * 8)] = (__bf16)v;
                }
            }
        }
        __syncthreads();   // msgT complete for all 512 stations

        // ========== GEMM2 (agg = adj @ msg) fused with GEMM3 (gates) + LSTM ==========
        for (int mt = 0; mt < 4; ++mt) {
            const int m0 = m0w + mt * 16;
            v8f agg[4];
            #pragma unroll
            for (int nt = 0; nt < 4; ++nt)
                #pragma unroll
                for (int r = 0; r < 8; ++r) agg[nt][r] = 0.0f;

            for (int ks = 0; ks < 16; ++ks) {   // K = 512
                v16bf A = load_tile16(adjb + (size_t)(m0 + lrow) * N_ST + ks * 32 + lhalf * 8);
                #pragma unroll
                for (int nt = 0; nt < 4; ++nt) {
                    v16bf B = load_tile16(msgT_s + (nt * 16 + lrow) * N_ST + ks * 32 + lhalf * 8);
                    agg[nt] = __builtin_amdgcn_wmma_f32_16x16x32_bf16(
                        false, A, false, B, (short)0, agg[nt], false, false);
                }
            }
            // C layout -> A layout via per-wave LDS scratch (bf16, [16][64])
            #pragma unroll
            for (int nt = 0; nt < 4; ++nt) {
                const int col = nt * 16 + lrow;
                #pragma unroll
                for (int r = 0; r < 8; ++r)
                    aggscr[(r + lhalf * 8) * HDIM + col] = (__bf16)agg[nt][r];
            }
            asm volatile("s_wait_dscnt 0" ::: "memory");   // own-wave LDS RAW

            v16bf Ag[2];
            #pragma unroll
            for (int ks = 0; ks < 2; ++ks)
                Ag[ks] = load_tile16(aggscr + lrow * HDIM + ks * 32 + lhalf * 8);

            #pragma unroll
            for (int j = 0; j < 4; ++j) {      // 16 gate columns at a time
                v8f gi, gf, gg, go;
                {
                    const float bi = bl_s[0 * HDIM + j * 16 + lrow];
                    const float bf = bl_s[1 * HDIM + j * 16 + lrow];
                    const float bG = bl_s[2 * HDIM + j * 16 + lrow];
                    const float bo = bl_s[3 * HDIM + j * 16 + lrow];
                    #pragma unroll
                    for (int r = 0; r < 8; ++r) { gi[r]=bi; gf[r]=bf; gg[r]=bG; go[r]=bo; }
                }
                #pragma unroll
                for (int ks = 0; ks < 2; ++ks) {
                    const int ko = ks * 32 + lhalf * 8;
                    v16bf Bi = load_tile16(WlT_s + (0 * HDIM + j * 16 + lrow) * HDIM + ko);
                    v16bf Bf = load_tile16(WlT_s + (1 * HDIM + j * 16 + lrow) * HDIM + ko);
                    v16bf Bg = load_tile16(WlT_s + (2 * HDIM + j * 16 + lrow) * HDIM + ko);
                    v16bf Bo = load_tile16(WlT_s + (3 * HDIM + j * 16 + lrow) * HDIM + ko);
                    gi = __builtin_amdgcn_wmma_f32_16x16x32_bf16(false, Ag[ks], false, Bi, (short)0, gi, false, false);
                    gf = __builtin_amdgcn_wmma_f32_16x16x32_bf16(false, Ag[ks], false, Bf, (short)0, gf, false, false);
                    gg = __builtin_amdgcn_wmma_f32_16x16x32_bf16(false, Ag[ks], false, Bg, (short)0, gg, false, false);
                    go = __builtin_amdgcn_wmma_f32_16x16x32_bf16(false, Ag[ks], false, Bo, (short)0, go, false, false);
                }
                // LSTM elementwise update (rows owned by this wave -> no cross-wave hazard)
                #pragma unroll
                for (int r = 0; r < 8; ++r) {
                    const int row = m0 + r + lhalf * 8;
                    const int col = j * 16 + lrow;
                    const int idx = row * HDIM + col;
                    const float iv = sigm(gi[r]);
                    const float fv = sigm(gf[r]);
                    const float gv = fast_tanh(gg[r]);
                    const float ov = sigm(go[r]);
                    const float cn = fv * c_s[idx] + iv * gv;
                    c_s[idx] = cn;
                    h_s[idx] = (__bf16)(ov * fast_tanh(cn));
                }
            }
        }
        __syncthreads();   // all waves done reading msgT / writing h before next step
    }

    // ---------------- readout: relu(h[b,0,:] @ Wr1 + br1) @ Wr2 + br2 ----------------
    if (wave == 0) {
        float acc = br1[lane];
        for (int k = 0; k < HDIM; ++k)
            acc += (float)h_s[k] * Wr1[k * 32 + lane];
        acc = acc > 0.0f ? acc : 0.0f;
        float p = acc * Wr2[lane];
        #pragma unroll
        for (int off = 16; off > 0; off >>= 1)
            p += __shfl_down(p, off, 32);
        if (lane == 0) out[b] = p + br2[0];
    }
}

// prepass: adj f32 -> bf16 into workspace (L2-resident thereafter)
__global__ void adj_to_bf16(const float* __restrict__ adj, __bf16* __restrict__ outb) {
    const int i = blockIdx.x * blockDim.x + threadIdx.x;
    outb[i] = (__bf16)adj[i];
}

extern "C" void kernel_launch(void* const* d_in, const int* in_sizes, int n_in,
                              void* d_out, int out_size, void* d_ws, size_t ws_size,
                              hipStream_t stream) {
    (void)in_sizes; (void)n_in; (void)out_size; (void)ws_size;
    const float* seq = (const float*)d_in[0];
    const float* adj = (const float*)d_in[1];
    const float* Wg  = (const float*)d_in[2];
    const float* bg  = (const float*)d_in[3];
    const float* Wl  = (const float*)d_in[4];
    const float* bl  = (const float*)d_in[5];
    const float* Wr1 = (const float*)d_in[6];
    const float* br1 = (const float*)d_in[7];
    const float* Wr2 = (const float*)d_in[8];
    const float* br2 = (const float*)d_in[9];

    __bf16* adjb = (__bf16*)d_ws;   // 512*512*2 = 512 KB

    adj_to_bf16<<<(N_ST * N_ST) / 256, 256, 0, stream>>>(adj, adjb);
    gclstm_kernel<<<32, 256, 0, stream>>>(seq, adjb, Wg, bg, Wl, bl,
                                          Wr1, br1, Wr2, br2, (float*)d_out);
}
